// RNN_BN_simple_25598005084673
// MI455X (gfx1250) — matrix operand order repaired and checked by reading, not compiled, not run on hardware
//
#include <hip/hip_runtime.h>

// Problem constants (match reference).
#define B_   4096
#define T_   64
#define P_   64
#define H1_  512
#define H2_  512
#define EPS_ 1e-3f

typedef __bf16 bf16_t;
typedef bf16_t v16bf __attribute__((ext_vector_type(16)));
typedef float  v8f   __attribute__((ext_vector_type(8)));
typedef unsigned int u32x4 __attribute__((ext_vector_type(4)));
typedef int          i32x4 __attribute__((ext_vector_type(4)));
typedef int          i32x8 __attribute__((ext_vector_type(8)));

// -------- LDS layout (static, 65,024 bytes total) --------------------------
// NOTE: single static __shared__ object -> LDS base 0; TDM descriptors use
// __builtin_offsetof(SharedMem, ...) as the raw LDS byte address.
struct SharedMem {
  bf16_t A[128][66];        // normalized input tile (128 rows x 64 k)  -> 16,896 B
  bf16_t Bt[512][34];       // weight chunk [n][k<=32], pad->34 via TDM -> 34,816 B
  bf16_t stag[8][16][36];   // per-wave D->A relayout staging           ->  9,216 B
  float  s_sum[512];        //                                          ->  2,048 B
  float  s_sq[512];         //                                          ->  2,048 B
};

// -------- WMMA fragment helpers (CDNA5 wave32 layouts, CDNA5 ISA) ---------
__device__ __forceinline__ v8f zero8() {
  v8f z = {0.f, 0.f, 0.f, 0.f, 0.f, 0.f, 0.f, 0.f};
  return z;
}

// A (16x32 bf16): lanes 0-15 hold row M=lane, K {0..7,16..23}; lanes 16-31
// hold row M=lane-16, K {8..15,24..31}. Element e -> K = klo + (e<8 ? e : e+8).
__device__ __forceinline__ v16bf load_a_frag(const bf16_t* base, int stride, int lane) {
  int r   = lane & 15;
  int klo = (lane < 16) ? 0 : 8;
  const bf16_t* row = base + r * stride + klo;
  v16bf a;
#pragma unroll
  for (int e = 0; e < 16; ++e) {
    int k = (e < 8) ? e : (e + 8);
    a[e] = row[k];
  }
  return a;
}

// B (32x16 bf16) from K-transposed LDS [n][k]: lane holds col N=lane&15,
// K {0..7,16..23} (lanes 0-15) or {8..15,24..31} (lanes 16-31), matching A.
__device__ __forceinline__ v16bf load_b_frag(const bf16_t* base, int stride, int lane) {
  int n    = lane & 15;
  int koff = (lane < 16) ? 0 : 8;
  const bf16_t* row = base + n * stride + koff;
  v16bf b;
#pragma unroll
  for (int e = 0; e < 16; ++e) b[e] = row[(e < 8) ? e : (e + 8)];
  return b;
}

__device__ __forceinline__ v8f wmma_bf16(v16bf a, v16bf b, v8f c) {
  return __builtin_amdgcn_wmma_f32_16x16x32_bf16(false, a, false, b, (short)0, c,
                                                 false, false);
}

// -------- Tensor Data Mover: 2-D bf16 tile (rows x 32) -> LDS --------------
// D# packed per CDNA5 ISA:
//   group0: count=1, lds_addr, global_addr[56:0], type=2
//   group1: data_size=2B, pad_enable (16 DW interval, 1 DW pad -> LDS row
//           stride 34 bf16), tensor 32 x rows, tile 32 x rows, dim0 stride = K.
__device__ __forceinline__ void tdm_load_tile(unsigned lds_off, const bf16_t* gptr,
                                              unsigned rows, unsigned k_stride) {
  unsigned long long ga = (unsigned long long)(uintptr_t)gptr;
  u32x4 g0;
  g0[0] = 1u;                                                   // count=1
  g0[1] = lds_off;                                              // lds_addr
  g0[2] = (unsigned)(ga & 0xffffffffu);                         // global_addr lo
  g0[3] = (unsigned)((ga >> 32) & 0x01ffffffu) | (2u << 30);    // addr hi | type=2
  i32x8 g1;
  g1[0] = (1 << 16) | (1 << 20) | (3 << 22);   // data_size=2B, pad_en, intvl=16DW, amt=1DW
  g1[1] = (int)(32u << 16);                    // tensor_dim0 = 32 (lo16 @ [63:48])
  g1[2] = (int)((rows & 0xffffu) << 16);       // tensor_dim0 hi=0, tensor_dim1 lo
  g1[3] = (int)(32u << 16);                    // tensor_dim1 hi=0, tile_dim0 = 32
  g1[4] = (int)(rows & 0xffffu);               // tile_dim1 = rows, tile_dim2 = 0
  g1[5] = (int)k_stride;                       // tensor_dim0_stride lo32
  g1[6] = 0;
  g1[7] = 0;
  i32x4 z4 = {0, 0, 0, 0};
#if defined(__clang_major__) && (__clang_major__ >= 23)
  i32x8 z8 = {0, 0, 0, 0, 0, 0, 0, 0};
  __builtin_amdgcn_tensor_load_to_lds(g0, g1, z4, z4, z8, 0);
#else
  __builtin_amdgcn_tensor_load_to_lds(g0, g1, z4, z4, 0);
#endif
}

// -------- Pass 0: pack + K-transpose weights to bf16 -----------------------
// src: [T][K][N] f32  ->  dst: [T][N][K] bf16
__global__ __launch_bounds__(256) void pack_wt_kernel(const float* __restrict__ src,
                                                      bf16_t* __restrict__ dst,
                                                      int K, int N, size_t total) {
  size_t idx = (size_t)blockIdx.x * 256 + threadIdx.x;
  if (idx >= total) return;
  int k = (int)(idx % K);
  size_t r = idx / K;
  int n = (int)(r % N);
  int t = (int)(r / N);
  dst[idx] = (bf16_t)src[((size_t)t * K + k) * N + n];
}

// -------- Pass 1: BN stats of prices -> affine (a0, c0) --------------------
__global__ __launch_bounds__(256) void stats0_kernel(const float* __restrict__ prices,
                                                     const float* __restrict__ g0,
                                                     const float* __restrict__ beta0,
                                                     float* __restrict__ a0,
                                                     float* __restrict__ c0) {
  __shared__ float rs[256], rq[256];
  const int t = blockIdx.x;
  const int p = threadIdx.x & 63;
  const int grp = threadIdx.x >> 6;   // 0..3
  float s = 0.f, q = 0.f;
  for (int b = grp; b < B_; b += 4) {
    float v = prices[((size_t)b * T_ + t) * P_ + p];
    s += v;
    q += v * v;
  }
  rs[threadIdx.x] = s;
  rq[threadIdx.x] = q;
  __syncthreads();
  if (grp == 0) {
#pragma unroll
    for (int j = 1; j < 4; ++j) {
      s += rs[p + 64 * j];
      q += rq[p + 64 * j];
    }
    float m   = s * (1.f / (float)B_);
    float var = q * (1.f / (float)B_) - m * m;   // biased variance
    float sc  = g0[t * P_ + p] * rsqrtf(var + EPS_);
    a0[t * P_ + p] = sc;
    c0[t * P_ + p] = beta0[t * P_ + p] - m * sc;
  }
}

// -------- Finalize sum/sumsq -> affine (a, c) ------------------------------
__global__ __launch_bounds__(256) void finalize_kernel(const float* __restrict__ sum,
                                                       const float* __restrict__ sq,
                                                       const float* __restrict__ g,
                                                       const float* __restrict__ beta,
                                                       float* __restrict__ a,
                                                       float* __restrict__ c, int n) {
  int i = blockIdx.x * blockDim.x + threadIdx.x;
  if (i >= n) return;
  float m   = sum[i] * (1.f / (float)B_);
  float var = sq[i] * (1.f / (float)B_) - m * m;
  float sc  = g[i] * rsqrtf(var + EPS_);
  a[i] = sc;
  c[i] = beta[i] - m * sc;
}

// -------- Fused per-(t, 128-row) pipeline ----------------------------------
// STAGE 1: GEMM1, accumulate sum/sumsq of h1.
// STAGE 2: GEMM1 -> BN1+ReLU -> GEMM2, accumulate sum/sumsq of h2.
// STAGE 3: GEMM1 -> BN1+ReLU -> GEMM2 -> BN2+ReLU -> GEMM3 -> out (+b3).
template <int STAGE>
__global__ __launch_bounds__(256) void fused_kernel(
    const float* __restrict__ prices,
    const bf16_t* __restrict__ wb1, const float* __restrict__ b1,
    const bf16_t* __restrict__ wb2, const float* __restrict__ b2,
    const bf16_t* __restrict__ wb3, const float* __restrict__ b3,
    const float* __restrict__ a0, const float* __restrict__ c0,
    const float* __restrict__ a1, const float* __restrict__ c1,
    const float* __restrict__ a2, const float* __restrict__ c2,
    float* __restrict__ sum_out, float* __restrict__ sq_out,
    float* __restrict__ out) {
  __shared__ SharedMem sm;
  const int t    = blockIdx.y;
  const int m0   = blockIdx.x * 128;
  const int tid  = threadIdx.x;
  const int lane = tid & 31;
  const int wave = tid >> 5;     // 8 wave32s
  const int rowB = wave * 16;    // this wave's 16-row strip
  const int n0   = lane & 15;
  const int rhalf = (lane < 16) ? 0 : 8;
  const unsigned bt_lds = (unsigned)__builtin_offsetof(SharedMem, Bt);

  // Per-wave TDM band issue: wave w DMAs rows [w*band, (w+1)*band) of the
  // [rows x 32] weight chunk into Bt with padded stride 34.
  auto tdm_chunk = [&](const bf16_t* wbase, int rows, int K, int kk) {
    int band = rows >> 3;  // 8 waves
    unsigned lds = bt_lds + (unsigned)(wave * band) * 34u * 2u;
    const bf16_t* g = wbase + (size_t)(wave * band) * K + (size_t)kk * 32;
    tdm_load_tile(lds, g, (unsigned)band, (unsigned)K);
  };

  // ---- kick off GEMM1 weight chunk 0 while we stage the input tile --------
  const bf16_t* w1t = wb1 + (size_t)t * H1_ * P_;   // [512][64] bf16
  tdm_chunk(w1t, H1_, P_, 0);

  // ---- Stage the BN0-normalized input tile [128 x 64] as bf16 -------------
  {
    int p  = tid & 63;
    int r0 = tid >> 6;           // 0..3
    float aa = a0[t * P_ + p];
    float cc = c0[t * P_ + p];
#pragma unroll
    for (int j = 0; j < 32; ++j) {
      int row = r0 + j * 4;
      float v = prices[((size_t)(m0 + row) * T_ + t) * P_ + p];
      sm.A[row][p] = (bf16_t)(v * aa + cc);
    }
  }
  if constexpr (STAGE != 3) {
    for (int i = tid; i < 512; i += 256) {
      sm.s_sum[i] = 0.f;
      sm.s_sq[i]  = 0.f;
    }
  }

  // ---- GEMM1: h1_strip[16 x 512] = Xbn[16 x 64] @ W1_t[64 x 512] ----------
  v8f acc1[32];
#pragma unroll
  for (int j = 0; j < 32; ++j) acc1[j] = zero8();

#pragma unroll
  for (int kk = 0; kk < 2; ++kk) {
    __builtin_amdgcn_s_wait_tensorcnt((short)0);
    __syncthreads();                       // Bt chunk kk + A tile visible
    v16bf af = load_a_frag(&sm.A[rowB][kk * 32], 66, lane);
#pragma unroll
    for (int j = 0; j < 32; ++j) {
      v16bf bf = load_b_frag(&sm.Bt[j * 16][0], 34, lane);
      acc1[j] = wmma_bf16(af, bf, acc1[j]);
    }
    __syncthreads();                       // all Bt reads done
    if (kk == 0) tdm_chunk(w1t, H1_, P_, 1);
  }

  if constexpr (STAGE == 1) {
    // stats of h1 = acc1 + b1
#pragma unroll
    for (int j = 0; j < 32; ++j) {
      int n = j * 16 + n0;
      float bias = b1[t * H1_ + n];
      float s = 0.f, q = 0.f;
#pragma unroll
      for (int r = 0; r < 8; ++r) {
        float v = acc1[j][r] + bias;
        s += v;
        q += v * v;
      }
      atomicAdd(&sm.s_sum[n], s);
      atomicAdd(&sm.s_sq[n], q);
    }
    __syncthreads();
    for (int n = tid; n < 512; n += 256) {
      atomicAdd(&sum_out[t * H1_ + n], sm.s_sum[n]);
      atomicAdd(&sq_out[t * H1_ + n], sm.s_sq[n]);
    }
    return;
  } else {
    // ---- BN1 + ReLU (folded affine), keep strip in registers --------------
    float h1p[256];
#pragma unroll
    for (int j = 0; j < 32; ++j) {
      int n = j * 16 + n0;
      float aa = a1[t * H1_ + n];
      float cc = aa * b1[t * H1_ + n] + c1[t * H1_ + n];
#pragma unroll
      for (int r = 0; r < 8; ++r) {
        float v = aa * acc1[j][r] + cc;
        h1p[j * 8 + r] = v > 0.f ? v : 0.f;
      }
    }

    // ---- GEMM2: h2_strip[16 x 512] = h1p[16 x 512] @ W2_t[512 x 512] ------
    const bf16_t* w2t = wb2 + (size_t)t * H2_ * H1_;  // [512][512] bf16
    v8f acc2[32];
#pragma unroll
    for (int j = 0; j < 32; ++j) acc2[j] = zero8();

    tdm_chunk(w2t, H2_, H1_, 0);
#pragma unroll
    for (int kk = 0; kk < 16; ++kk) {
      {  // D-layout -> A-layout relayout for h1p cols [kk*32, kk*32+32)
#pragma unroll
        for (int tt = 0; tt < 2; ++tt) {
          int j = kk * 2 + tt;
#pragma unroll
          for (int r = 0; r < 8; ++r)
            sm.stag[wave][rhalf + r][tt * 16 + n0] = (bf16_t)h1p[j * 8 + r];
        }
      }
      __builtin_amdgcn_s_wait_tensorcnt((short)0);
      __syncthreads();
      v16bf af = load_a_frag(&sm.stag[wave][0][0], 36, lane);
#pragma unroll
      for (int j = 0; j < 32; ++j) {
        v16bf bf = load_b_frag(&sm.Bt[j * 16][0], 34, lane);
        acc2[j] = wmma_bf16(af, bf, acc2[j]);
      }
      __syncthreads();
      if (kk + 1 < 16) tdm_chunk(w2t, H2_, H1_, kk + 1);
    }

    if constexpr (STAGE == 2) {
      // stats of h2 = acc2 + b2
#pragma unroll
      for (int j = 0; j < 32; ++j) {
        int n = j * 16 + n0;
        float bias = b2[t * H2_ + n];
        float s = 0.f, q = 0.f;
#pragma unroll
        for (int r = 0; r < 8; ++r) {
          float v = acc2[j][r] + bias;
          s += v;
          q += v * v;
        }
        atomicAdd(&sm.s_sum[n], s);
        atomicAdd(&sm.s_sq[n], q);
      }
      __syncthreads();
      for (int n = tid; n < 512; n += 256) {
        atomicAdd(&sum_out[t * H2_ + n], sm.s_sum[n]);
        atomicAdd(&sq_out[t * H2_ + n], sm.s_sq[n]);
      }
      return;
    } else {
      // ---- BN2 + ReLU ------------------------------------------------------
      float h2p[256];
#pragma unroll
      for (int j = 0; j < 32; ++j) {
        int n = j * 16 + n0;
        float aa = a2[t * H2_ + n];
        float cc = aa * b2[t * H2_ + n] + c2[t * H2_ + n];
#pragma unroll
        for (int r = 0; r < 8; ++r) {
          float v = aa * acc2[j][r] + cc;
          h2p[j * 8 + r] = v > 0.f ? v : 0.f;
        }
      }

      // ---- GEMM3: out_strip[16 x 64] = h2p[16 x 512] @ W3_t[512 x 64] -----
      const bf16_t* w3t = wb3 + (size_t)t * P_ * H2_;  // [64][512] bf16
      v8f acc3[4];
#pragma unroll
      for (int j = 0; j < 4; ++j) acc3[j] = zero8();

      tdm_chunk(w3t, P_, H2_, 0);
#pragma unroll
      for (int kk = 0; kk < 16; ++kk) {
        {
#pragma unroll
          for (int tt = 0; tt < 2; ++tt) {
            int j = kk * 2 + tt;
#pragma unroll
            for (int r = 0; r < 8; ++r)
              sm.stag[wave][rhalf + r][tt * 16 + n0] = (bf16_t)h2p[j * 8 + r];
          }
        }
        __builtin_amdgcn_s_wait_tensorcnt((short)0);
        __syncthreads();
        v16bf af = load_a_frag(&sm.stag[wave][0][0], 36, lane);
#pragma unroll
        for (int j = 0; j < 4; ++j) {
          v16bf bf = load_b_frag(&sm.Bt[j * 16][0], 34, lane);
          acc3[j] = wmma_bf16(af, bf, acc3[j]);
        }
        __syncthreads();
        if (kk + 1 < 16) tdm_chunk(w3t, P_, H2_, kk + 1);
      }

      // ---- epilogue: out = acc3 + b3 ---------------------------------------
#pragma unroll
      for (int j = 0; j < 4; ++j) {
        int p = j * 16 + n0;
        float bias = b3[t * P_ + p];
#pragma unroll
        for (int r = 0; r < 8; ++r) {
          int brow = m0 + rowB + rhalf + r;
          out[((size_t)brow * T_ + t) * P_ + p] = acc3[j][r] + bias;
        }
      }
    }
  }
}

// ---------------------------------------------------------------------------
extern "C" void kernel_launch(void* const* d_in, const int* in_sizes, int n_in,
                              void* d_out, int out_size, void* d_ws, size_t ws_size,
                              hipStream_t stream) {
  (void)in_sizes; (void)n_in; (void)out_size; (void)ws_size;
  const float* prices = (const float*)d_in[0];
  const float* bn0_g  = (const float*)d_in[1];
  const float* bn0_b  = (const float*)d_in[2];
  const float* W1     = (const float*)d_in[3];
  const float* b1     = (const float*)d_in[4];
  const float* bn1_g  = (const float*)d_in[5];
  const float* bn1_b  = (const float*)d_in[6];
  const float* W2     = (const float*)d_in[7];
  const float* b2     = (const float*)d_in[8];
  const float* bn2_g  = (const float*)d_in[9];
  const float* bn2_b  = (const float*)d_in[10];
  const float* W3     = (const float*)d_in[11];
  const float* b3     = (const float*)d_in[12];
  float* out = (float*)d_out;

  // Workspace layout. Float stats region (~1.03 MB), then bf16 weights (~40 MB).
  float* ws   = (float*)d_ws;
  float* a0   = ws;               // T*P
  float* c0   = ws + 4096;        // T*P
  float* sum1 = ws + 8192;        // T*H1
  float* sq1  = ws + 40960;       // T*H1
  float* sum2 = ws + 73728;       // T*H2
  float* sq2  = ws + 106496;      // T*H2
  float* a1   = ws + 139264;
  float* c1   = ws + 172032;
  float* a2   = ws + 204800;
  float* c2   = ws + 237568;
  bf16_t* wb1 = (bf16_t*)((char*)d_ws + 1081344);              // [T][H1][P]
  bf16_t* wb2 = (bf16_t*)((char*)d_ws + 1081344 + 4194304);    // [T][H2][H1]
  bf16_t* wb3 = (bf16_t*)((char*)d_ws + 1081344 + 37748736);   // [T][P][H2]

  // Zero the four stats accumulators (contiguous 4 * 32768 floats).
  hipMemsetAsync(sum1, 0, (size_t)4 * 32768 * sizeof(float), stream);

  // Pass 0: pack + transpose weights to bf16 [T][N][K].
  {
    size_t n1 = (size_t)T_ * H1_ * P_;   // 2,097,152
    size_t n2 = (size_t)T_ * H2_ * H1_;  // 16,777,216
    size_t n3 = (size_t)T_ * P_ * H2_;   // 2,097,152
    pack_wt_kernel<<<(unsigned)(n1 / 256), 256, 0, stream>>>(W1, wb1, P_, H1_, n1);
    pack_wt_kernel<<<(unsigned)(n2 / 256), 256, 0, stream>>>(W2, wb2, H1_, H2_, n2);
    pack_wt_kernel<<<(unsigned)(n3 / 256), 256, 0, stream>>>(W3, wb3, H2_, P_, n3);
  }

  dim3 grid(B_ / 128, T_);  // (32, 64)

  // Pass 1: BN0 stats.
  stats0_kernel<<<T_, 256, 0, stream>>>(prices, bn0_g, bn0_b, a0, c0);

  // Pass 2: GEMM1 + h1 stats.
  fused_kernel<1><<<grid, 256, 0, stream>>>(prices, wb1, b1, wb2, b2, wb3, b3,
                                            a0, c0, a1, c1, a2, c2,
                                            sum1, sq1, nullptr);
  finalize_kernel<<<(T_ * H1_) / 256, 256, 0, stream>>>(sum1, sq1, bn1_g, bn1_b,
                                                        a1, c1, T_ * H1_);

  // Pass 3: GEMM1 -> GEMM2 + h2 stats.
  fused_kernel<2><<<grid, 256, 0, stream>>>(prices, wb1, b1, wb2, b2, wb3, b3,
                                            a0, c0, a1, c1, a2, c2,
                                            sum2, sq2, nullptr);
  finalize_kernel<<<(T_ * H2_) / 256, 256, 0, stream>>>(sum2, sq2, bn2_g, bn2_b,
                                                        a2, c2, T_ * H2_);

  // Pass 4: full chain -> output.
  fused_kernel<3><<<grid, 256, 0, stream>>>(prices, wb1, b1, wb2, b2, wb3, b3,
                                            a0, c0, a1, c1, a2, c2,
                                            nullptr, nullptr, out);
}
